// DepthSeparableConv2d_conv2_4_37572373905415
// MI455X (gfx1250) — compile-verified, matrix operand order checked by reading, not run on hardware
//
#include <hip/hip_runtime.h>
#include <hip/hip_bf16.h>

typedef __attribute__((ext_vector_type(16))) __bf16 v16bf;   // 8 VGPRs
typedef __attribute__((ext_vector_type(8)))  float  v8f;     // 8 VGPRs

#define BATCH 32
#define CIN   256
#define COUT  512
#define HIN   56
#define WIN   56
#define HOUT  54
#define WOUT  54
#define HWI   (HIN * WIN)    // 3136
#define HWO   (HOUT * WOUT)  // 2916
#define EPSV  1e-5f

// Split fp32 into two bf16 words (x ~= hi + lo), pack {hi,lo} into one dword.
__device__ __forceinline__ unsigned pack_split(float v) {
    __bf16 h = (__bf16)v;
    __bf16 l = (__bf16)(v - (float)h);
    return ((unsigned)__builtin_bit_cast(unsigned short, h) << 16) |
           (unsigned)__builtin_bit_cast(unsigned short, l);
}

// Load one 16-element bf16 fragment: two contiguous 8-element runs
// (ISA 7.12.2 16-bit layout: lanes 0-15 -> K {0..7,16..23}; lanes 16-31 ->
//  K {8..15,24..31}). row points at 40-bf16 (80B, 16B-aligned) LDS row.
__device__ __forceinline__ v16bf load_frag(const __bf16* row, int hi) {
    union { v16bf v; uint4 q[2]; } u;
    u.q[0] = *(const uint4*)(row + hi * 8);
    u.q[1] = *(const uint4*)(row + hi * 8 + 16);
    return u.v;
}

// ---------------------------------------------------------------------------
// Kernel 0: zero the per-(b,o) running-max buffer.
// ---------------------------------------------------------------------------
__global__ __launch_bounds__(256) void init_max_kernel(float* __restrict__ maxbuf) {
    int i = blockIdx.x * 256 + threadIdx.x;
    if (i < BATCH * COUT) maxbuf[i] = 0.0f;
}

// ---------------------------------------------------------------------------
// Kernel 0b: pre-split pointwise weights into packed {hi,lo} bf16 dwords.
// ---------------------------------------------------------------------------
__global__ __launch_bounds__(256) void pack_w_kernel(
    const float* __restrict__ pw_w, unsigned* __restrict__ wpk) {
    int i = blockIdx.x * 256 + threadIdx.x;
    if (i < COUT * CIN) wpk[i] = pack_split(pw_w[i]);
}

// ---------------------------------------------------------------------------
// Kernel 1: depthwise 3x3 VALID conv + bias + BN + ReLU + _cut(4.0).
// One block per (b,c) map; writes y as packed {hi,lo} bf16 dwords.
// ---------------------------------------------------------------------------
__global__ __launch_bounds__(256) void dw_kernel(
    const float* __restrict__ x, const float* __restrict__ dw_w,
    const float* __restrict__ dw_b, const float* __restrict__ dw_gamma,
    const float* __restrict__ dw_beta, const float* __restrict__ dw_mean,
    const float* __restrict__ dw_var, unsigned* __restrict__ ypk)
{
    const int bc = blockIdx.x;         // b*CIN + c
    const int c  = bc & (CIN - 1);
    const int tid = threadIdx.x;

    __shared__ float tile[HWI];
    __shared__ float red[256];

    const float* xin = x + (size_t)bc * HWI;
    for (int i = tid; i < HWI; i += 256) tile[i] = xin[i];

    float w[9];
#pragma unroll
    for (int i = 0; i < 9; ++i) w[i] = dw_w[c * 9 + i];
    const float scale = dw_gamma[c] * rsqrtf(dw_var[c] + EPSV);
    const float shift = dw_beta[c] + (dw_b[c] - dw_mean[c]) * scale;

    __syncthreads();

    float vals[12];
    float lmax = 0.0f;
#pragma unroll
    for (int it = 0; it < 12; ++it) {
        const int idx = tid + it * 256;
        float v = 0.0f;
        if (idx < HWO) {
            const int oh = idx / WOUT;
            const int ow = idx - oh * WOUT;
            float acc = 0.0f;
#pragma unroll
            for (int kh = 0; kh < 3; ++kh)
#pragma unroll
                for (int kw = 0; kw < 3; ++kw)
                    acc = fmaf(tile[(oh + kh) * WIN + ow + kw], w[kh * 3 + kw], acc);
            v = fmaxf(fmaf(acc, scale, shift), 0.0f);
        }
        vals[it] = v;
        lmax = fmaxf(lmax, v);
    }

    red[tid] = lmax;
    __syncthreads();
    for (int s = 128; s > 0; s >>= 1) {
        if (tid < s) red[tid] = fmaxf(red[tid], red[tid + s]);
        __syncthreads();
    }
    const bool cut = red[0] < 4.0f;

    unsigned* yo = ypk + (size_t)bc * HWO;
#pragma unroll
    for (int it = 0; it < 12; ++it) {
        const int idx = tid + it * 256;
        if (idx < HWO) yo[idx] = pack_split(cut ? 0.0f : vals[it]);
    }
}

// ---------------------------------------------------------------------------
// Kernel 2: pointwise GEMM via bf16x3 split on V_WMMA_F32_16X16X32_BF16.
// Block tile 64(M) x 128(N); K-chunk = 32 = one WMMA K-step; 8 waves
// (4 in M x 2 in N), each wave owns a 16x64 strip (4 x v8f accumulators).
// LDS: A [m][k] and B transposed [n][k], rows padded to 40 bf16 so every
// fragment is two 16B-aligned ds_load_b128 with conflict-free banking.
// ---------------------------------------------------------------------------
__global__ __launch_bounds__(256) void pw_gemm_kernel(
    const unsigned* __restrict__ ypk, const unsigned* __restrict__ wpk,
    const float* __restrict__ pw_b, const float* __restrict__ pw_gamma,
    const float* __restrict__ pw_beta, const float* __restrict__ pw_mean,
    const float* __restrict__ pw_var, float* __restrict__ z,
    float* __restrict__ maxbuf)
{
    const int b  = blockIdx.z;
    const int m0 = blockIdx.y * 64;
    const int n0 = blockIdx.x * 128;
    const int tid  = threadIdx.x;
    const int wave = tid >> 5;
    const int lane = tid & 31;
    const int wm = (wave & 3) * 16;
    const int wn = (wave >> 2) * 64;
    const int nl = lane & 15;
    const int hi = lane >> 4;

    __shared__ __align__(16) __bf16 Ah[64][40];
    __shared__ __align__(16) __bf16 Al[64][40];
    __shared__ __align__(16) __bf16 Bh[128][40];
    __shared__ __align__(16) __bf16 Bl[128][40];
    __shared__ unsigned rowmax_s[64];

    if (tid < 64) rowmax_s[tid] = 0u;

    v8f acc[4] = {};
    const unsigned* yb = ypk + (size_t)b * CIN * HWO;

    // per-thread staging roles (fixed across K loop)
    const int am  = tid >> 2;             // A: m row 0..63
    const int ako = (tid & 3) * 8;        // A: k offset 0,8,16,24
    const int bnl = tid & 127;            // B: local n column
    const int bkh = (tid >> 7) * 16;      // B: k half 0 or 16
    const int bn  = n0 + bnl;

    for (int k0 = 0; k0 < CIN; k0 += 32) {
        // ---- stage A (unpack packed weights into hi/lo planes) ----
        {
            const unsigned* src = wpk + (size_t)(m0 + am) * CIN + k0 + ako;
            unsigned h[4], l[4];
#pragma unroll
            for (int i = 0; i < 4; ++i) {
                const unsigned p0 = src[2 * i];
                const unsigned p1 = src[2 * i + 1];
                h[i] = (p0 >> 16) | (p1 & 0xffff0000u);   // {hi_k, hi_k+1}
                l[i] = (p0 & 0xffffu) | (p1 << 16);       // {lo_k, lo_k+1}
            }
            uint4 th = {h[0], h[1], h[2], h[3]};
            uint4 tl = {l[0], l[1], l[2], l[3]};
            *(uint4*)&Ah[am][ako] = th;
            *(uint4*)&Al[am][ako] = tl;
        }
        // ---- stage B with transpose: global [k][n] -> LDS [n][k] ----
        {
            const unsigned* src = yb + (size_t)(k0 + bkh) * HWO + bn;
            unsigned h[8], l[8];
#pragma unroll
            for (int i = 0; i < 8; ++i) {
                unsigned p0 = 0u, p1 = 0u;
                if (bn < HWO) {
                    p0 = src[(size_t)(2 * i) * HWO];
                    p1 = src[(size_t)(2 * i + 1) * HWO];
                }
                h[i] = (p0 >> 16) | (p1 & 0xffff0000u);
                l[i] = (p0 & 0xffffu) | (p1 << 16);
            }
            uint4 th0 = {h[0], h[1], h[2], h[3]};
            uint4 th1 = {h[4], h[5], h[6], h[7]};
            uint4 tl0 = {l[0], l[1], l[2], l[3]};
            uint4 tl1 = {l[4], l[5], l[6], l[7]};
            *(uint4*)&Bh[bnl][bkh]     = th0;
            *(uint4*)&Bh[bnl][bkh + 8] = th1;
            *(uint4*)&Bl[bnl][bkh]     = tl0;
            *(uint4*)&Bl[bnl][bkh + 8] = tl1;
        }
        // prefetch next K-chunk of y (global_prefetch_b8)
        if (k0 + 32 < CIN && bn < HWO) {
            __builtin_prefetch(yb + (size_t)(k0 + 32 + bkh) * HWO + bn, 0, 3);
        }
        __syncthreads();

        // ---- compute: bf16x3 -> 12 WMMAs per chunk per wave ----
        const v16bf ah = load_frag(&Ah[wm + nl][0], hi);
        const v16bf al = load_frag(&Al[wm + nl][0], hi);
#pragma unroll
        for (int t = 0; t < 4; ++t) {
            const __bf16* brow_h = &Bh[wn + t * 16 + nl][0];
            const __bf16* brow_l = &Bl[wn + t * 16 + nl][0];
            const v16bf bh = load_frag(brow_h, hi);
            const v16bf bl = load_frag(brow_l, hi);
            acc[t] = __builtin_amdgcn_wmma_f32_16x16x32_bf16(
                false, ah, false, bh, (short)0, acc[t], false, false);
            acc[t] = __builtin_amdgcn_wmma_f32_16x16x32_bf16(
                false, ah, false, bl, (short)0, acc[t], false, false);
            acc[t] = __builtin_amdgcn_wmma_f32_16x16x32_bf16(
                false, al, false, bh, (short)0, acc[t], false, false);
        }
        __syncthreads();
    }

    // ---- epilogue: bias + BN + ReLU, store, per-row max for PW cut ----
#pragma unroll
    for (int r = 0; r < 8; ++r) {
        const int o = m0 + wm + hi * 8 + r;
        const float s  = pw_gamma[o] * rsqrtf(pw_var[o] + EPSV);
        const float sh = pw_beta[o] + (pw_b[o] - pw_mean[o]) * s;
        float tmax = 0.0f;
#pragma unroll
        for (int t = 0; t < 4; ++t) {
            const int n = n0 + wn + t * 16 + nl;
            float v = fmaxf(fmaf(acc[t][r], s, sh), 0.0f);
            if (n < HWO) {
                z[((size_t)b * COUT + o) * HWO + n] = v;
            } else {
                v = 0.0f;
            }
            tmax = fmaxf(tmax, v);
        }
        atomicMax(&rowmax_s[wm + hi * 8 + r], __float_as_uint(tmax));
    }
    __syncthreads();
    if (tid < 64) {
        atomicMax((unsigned*)&maxbuf[b * COUT + m0 + tid], rowmax_s[tid]);
    }
}

// ---------------------------------------------------------------------------
// Kernel 3: pointwise _cut(0.001) — zero only failing (b,o) maps.
// ---------------------------------------------------------------------------
__global__ __launch_bounds__(256) void pw_cut_kernel(
    const float* __restrict__ maxbuf, float* __restrict__ z)
{
    const int bo = blockIdx.y;
    if (maxbuf[bo] < 0.001f) {
        const int i = blockIdx.x * 256 + threadIdx.x;
        if (i < HWO) z[(size_t)bo * HWO + i] = 0.0f;
    }
}

// ---------------------------------------------------------------------------
extern "C" void kernel_launch(void* const* d_in, const int* in_sizes, int n_in,
                              void* d_out, int out_size, void* d_ws, size_t ws_size,
                              hipStream_t stream) {
    const float* x        = (const float*)d_in[0];
    const float* dw_w     = (const float*)d_in[1];
    const float* dw_b     = (const float*)d_in[2];
    const float* dw_gamma = (const float*)d_in[3];
    const float* dw_beta  = (const float*)d_in[4];
    const float* dw_mean  = (const float*)d_in[5];
    const float* dw_var   = (const float*)d_in[6];
    const float* pw_w     = (const float*)d_in[7];
    const float* pw_b     = (const float*)d_in[8];
    const float* pw_gamma = (const float*)d_in[9];
    const float* pw_beta  = (const float*)d_in[10];
    const float* pw_mean  = (const float*)d_in[11];
    const float* pw_var   = (const float*)d_in[12];

    float* z = (float*)d_out;

    // workspace: ypk (B*CIN*HWO dwords) | maxbuf (B*COUT f32) | wpk (COUT*CIN dwords)
    unsigned* ypk = (unsigned*)d_ws;
    char* p = (char*)d_ws + (size_t)BATCH * CIN * HWO * sizeof(unsigned);
    float* maxbuf = (float*)p;
    unsigned* wpk = (unsigned*)(p + (size_t)BATCH * COUT * sizeof(float));

    init_max_kernel<<<(BATCH * COUT + 255) / 256, 256, 0, stream>>>(maxbuf);
    pack_w_kernel<<<(COUT * CIN + 255) / 256, 256, 0, stream>>>(pw_w, wpk);

    dw_kernel<<<BATCH * CIN, 256, 0, stream>>>(
        x, dw_w, dw_b, dw_gamma, dw_beta, dw_mean, dw_var, ypk);

    dim3 g2((HWO + 127) / 128, COUT / 64, BATCH);   // (23, 8, 32)
    pw_gemm_kernel<<<g2, 256, 0, stream>>>(
        ypk, wpk, pw_b, pw_gamma, pw_beta, pw_mean, pw_var, z, maxbuf);

    dim3 g3((HWO + 255) / 256, BATCH * COUT);       // (12, 16384)
    pw_cut_kernel<<<g3, 256, 0, stream>>>(maxbuf, z);
}
